// CNNLSTM_24094766531257
// MI455X (gfx1250) — matrix operand order repaired and checked by reading, not compiled, not run on hardware
//
#include <hip/hip_runtime.h>
#include <hip/hip_bf16.h>
#include <math.h>

// ---------------- types / constants ----------------
typedef __attribute__((ext_vector_type(16))) __bf16 v16bf;
typedef __attribute__((ext_vector_type(8)))  float  v8f;

union Frag16 { unsigned int u[8]; v16bf v; };

#define B_   16
#define D0_  161
#define T0_  600
#define D1_  81
#define TC_  300
#define C_   32
#define D2_  41
#define F1_  1312      // 32*41
#define H_   768
#define G4_  3072      // 4*H
#define K2_  7392      // 21*11*32
#define NWG_ 24        // LSTM WGs per direction: 768/32
#define MROWS_ (TC_*B_) // 4800

// ---------------- small helpers ----------------
__device__ __forceinline__ unsigned short f2bf(float f) {
    unsigned x = __float_as_uint(f);
    return (unsigned short)((x + 0x7fffu + ((x >> 16) & 1u)) >> 16);
}
__device__ __forceinline__ float sigmf(float x) { return 1.f / (1.f + expf(-x)); }

// CDNA5 async global->LDS copy (ASYNCcnt-tracked) + wait
__device__ __forceinline__ void async_load_b128(unsigned lds_off, const void* gaddr) {
    unsigned long long a = (unsigned long long)gaddr;
    asm volatile("global_load_async_to_lds_b128 %0, %1, off" :: "v"(lds_off), "v"(a) : "memory");
}
__device__ __forceinline__ void wait_async0() {
    asm volatile("s_wait_asynccnt 0x0" ::: "memory");
}

// A fragment: 16x32 (MxK) bf16, row-major source, lda even.
__device__ __forceinline__ v16bf load_a_frag(const unsigned short* A, int lda, int lane) {
    Frag16 fr;
    int m = lane & 15, half = lane >> 4;
#pragma unroll
    for (int r = 0; r < 8; ++r) {
        int k = ((r & 3) * 2) + ((r >> 2) * 16) + half * 8;
        fr.u[r] = *(const unsigned int*)(A + (size_t)m * lda + k);
    }
    return fr.v;
}
// B fragment: 32x16 (KxN) bf16 from N-major packed (Bt: N rows x K cols), ldb even.
__device__ __forceinline__ v16bf load_b_frag(const unsigned short* Bt, int ldb, int lane) {
    Frag16 fr;
    int n = lane & 15, half = lane >> 4;
#pragma unroll
    for (int r = 0; r < 8; ++r) {
        int k = ((r & 3) * 2) + ((r >> 2) * 16) + half * 8;
        fr.u[r] = *(const unsigned int*)(Bt + (size_t)n * ldb + k);
    }
    return fr.v;
}

// ---------------- init: zero counters/stats, sum of out_len ----------------
__global__ void ds2_init_kernel(int* cnt, float* stats, int* sum_ol, const int* lengths) {
    int tid = threadIdx.x;
    for (int i = tid; i < 6000; i += 256) cnt[i] = 0;
    for (int i = tid; i < 128; i += 256) stats[i] = 0.f;
    if (tid == 0) {
        int s = 0;
        for (int b = 0; b < B_; ++b) s += (lengths[b] - 1) / 2 + 1;
        *sum_ol = s;
    }
}

// ---------------- weight transpose-pack f32(K,N) -> bf16(Npad,K) ----------------
__global__ void ds2_packT_kernel(const float* in, unsigned short* out, int K, int N, int Npad) {
    size_t idx = (size_t)blockIdx.x * blockDim.x + threadIdx.x;
    size_t total = (size_t)Npad * K;
    if (idx >= total) return;
    int n = (int)(idx / K), k = (int)(idx % K);
    out[idx] = (n < N) ? f2bf(in[(size_t)k * N + n]) : (unsigned short)0;
}

// ---------------- conv1 direct (C_in = 1) ----------------
__global__ void ds2_conv1_kernel(const float* __restrict__ x, const float* __restrict__ W,
                                 const float* __restrict__ bias, float* __restrict__ out) {
    int idx = blockIdx.x * blockDim.x + threadIdx.x;
    const int total = B_ * D1_ * TC_ * C_;
    if (idx >= total) return;
    int c = idx & 31;
    int t = (idx >> 5) % TC_;
    int d = ((idx >> 5) / TC_) % D1_;
    int b = idx / (C_ * TC_ * D1_);
    float acc = bias[c];
    int d0 = 2 * d - 20, t0 = 2 * t - 5;
    for (int kd = 0; kd < 41; ++kd) {
        int di = d0 + kd;
        if ((unsigned)di >= (unsigned)D0_) continue;
        const float* xr = x + ((size_t)b * D0_ + di) * T0_;
        const float* wr = W + (size_t)kd * 11 * 32 + c;
        for (int kt = 0; kt < 11; ++kt) {
            int ti = t0 + kt;
            if ((unsigned)ti < (unsigned)T0_) acc = fmaf(xr[ti], wr[kt * 32], acc);
        }
    }
    out[idx] = acc;
}

// ---------------- per-channel sum / sumsq (channel-last, 32 channels) ----------------
__global__ void ds2_bn_stats_kernel(const float* __restrict__ x, float* stat, long long n) {
    __shared__ float ss[64];
    if (threadIdx.x < 64) ss[threadIdx.x] = 0.f;
    __syncthreads();
    int c = threadIdx.x & 31;
    float s = 0.f, q = 0.f;
    for (long long i = (long long)blockIdx.x * blockDim.x + threadIdx.x; i < n;
         i += (long long)gridDim.x * blockDim.x) {
        float v = x[i]; s += v; q += v * v;
    }
    atomicAdd(&ss[c], s);
    atomicAdd(&ss[32 + c], q);
    __syncthreads();
    if (threadIdx.x < 64) atomicAdd(&stat[threadIdx.x], ss[threadIdx.x]);
}

// ---------------- BN1 apply + clip + mask -> bf16 NHWC ----------------
__global__ void ds2_bn1_apply_kernel(const float* __restrict__ x, const float* __restrict__ stat,
                                     const float* __restrict__ scale, const float* __restrict__ bias,
                                     const int* __restrict__ lengths, const int* __restrict__ sum_ol,
                                     unsigned short* __restrict__ out) {
    int idx = blockIdx.x * blockDim.x + threadIdx.x;
    const int total = B_ * D1_ * TC_ * C_;
    if (idx >= total) return;
    int c = idx & 31;
    int t = (idx >> 5) % TC_;
    int b = idx / (C_ * TC_ * D1_);
    float nz = (float)(D1_ * (*sum_ol));
    float mean = stat[c] / nz;
    float var = fmaxf(0.f, stat[32 + c] / nz - mean * mean);
    float v = (x[idx] - mean) * rsqrtf(var + 1e-5f) * scale[c] + bias[c];
    v = fminf(fmaxf(v, 0.f), 20.f);
    int ol = (lengths[b] - 1) / 2 + 1;
    if (t >= ol) v = 0.f;
    out[idx] = f2bf(v);
}

// ---------------- conv2 implicit-GEMM WMMA: (B,81,300,32)bf16 -> (B,41,300,32)f32 ----------------
// Branch-free gather: clamp address, select-zero value (no exec-mask divergence).
__device__ __forceinline__ unsigned conv2_fetch(const unsigned short* __restrict__ xin,
                                                int b, int d0, int t0, int kk) {
    int kd = kk / 352;
    int r2 = kk - kd * 352;
    int kt = r2 >> 5, ci = r2 & 31;
    int di = d0 + kd, ti = t0 + kt;
    bool ok = ((unsigned)di < (unsigned)D1_) && ((unsigned)ti < (unsigned)TC_);
    int dic = min(max(di, 0), D1_ - 1);
    int tic = min(max(ti, 0), TC_ - 1);
    unsigned v = (unsigned)xin[(((size_t)b * D1_ + dic) * TC_ + tic) * C_ + ci];
    return ok ? v : 0u;
}

__global__ void ds2_conv2_wmma_kernel(const unsigned short* __restrict__ xin,
                                      const unsigned short* __restrict__ Wt, // (32, 7392)
                                      const float* __restrict__ bias, float* __restrict__ out) {
    const int pixTiles = (B_ * D2_ * TC_) / 16; // 12300
    int wave = threadIdx.x >> 5, lane = threadIdx.x & 31;
    int pt = blockIdx.x * 8 + wave;             // one wave: 16 pixels x all 32 channels
    if (pt >= pixTiles) return;
    int m = lane & 15, half = lane >> 4;
    int p = pt * 16 + m;
    int b = p / (D2_ * TC_);
    int rem = p % (D2_ * TC_);
    int d = rem / TC_, t = rem % TC_;
    int d0 = 2 * d - 10, t0 = t - 5;
    v8f acc0 = {}, acc1 = {};
    for (int k0 = 0; k0 < K2_; k0 += 32) {
        Frag16 fa;
#pragma unroll
        for (int r = 0; r < 8; ++r) {
            int kk = k0 + ((r & 3) * 2) + ((r >> 2) * 16) + half * 8;
            unsigned lo = conv2_fetch(xin, b, d0, t0, kk);
            unsigned hi = conv2_fetch(xin, b, d0, t0, kk + 1);
            fa.u[r] = lo | (hi << 16);
        }
        v16bf fb0 = load_b_frag(Wt + k0, K2_, lane);
        v16bf fb1 = load_b_frag(Wt + (size_t)16 * K2_ + k0, K2_, lane);
        acc0 = __builtin_amdgcn_wmma_f32_16x16x32_bf16(false, fa.v, false, fb0, (short)0, acc0, false, false);
        acc1 = __builtin_amdgcn_wmma_f32_16x16x32_bf16(false, fa.v, false, fb1, (short)0, acc1, false, false);
    }
    int n = lane & 15;
#pragma unroll
    for (int r = 0; r < 8; ++r) {
        int prow = pt * 16 + r + 8 * half;
        out[(size_t)prow * 32 + n]      = acc0[r] + bias[n];
        out[(size_t)prow * 32 + 16 + n] = acc1[r] + bias[16 + n];
    }
}

// ---------------- BN2 apply + clip + mask -> feats bf16 (T,B,1312) ----------------
__global__ void ds2_bn2_apply_kernel(const float* __restrict__ x, const float* __restrict__ stat,
                                     const float* __restrict__ scale, const float* __restrict__ bias,
                                     const int* __restrict__ lengths, const int* __restrict__ sum_ol,
                                     unsigned short* __restrict__ feats) {
    int idx = blockIdx.x * blockDim.x + threadIdx.x;
    const int total = B_ * D2_ * TC_ * C_;
    if (idx >= total) return;
    int c = idx & 31;
    int t = (idx >> 5) % TC_;
    int d = ((idx >> 5) / TC_) % D2_;
    int b = idx / (C_ * TC_ * D2_);
    float nz = (float)(D2_ * (*sum_ol));
    float mean = stat[c] / nz;
    float var = fmaxf(0.f, stat[32 + c] / nz - mean * mean);
    float v = (x[idx] - mean) * rsqrtf(var + 1e-5f) * scale[c] + bias[c];
    v = fminf(fmaxf(v, 0.f), 20.f);
    int ol = (lengths[b] - 1) / 2 + 1;
    if (t >= ol) v = 0.f;
    feats[((size_t)(t * B_ + b)) * F1_ + c * D2_ + d] = f2bf(v);
}

// ---------------- length-aware time flip (bf16 rows (T,B,F)) ----------------
__global__ void ds2_flip_kernel(const unsigned short* __restrict__ in, unsigned short* __restrict__ out,
                                const int* __restrict__ lengths, int F) {
    size_t idx = (size_t)blockIdx.x * blockDim.x + threadIdx.x;
    size_t total = (size_t)TC_ * B_ * F;
    if (idx >= total) return;
    int f = (int)(idx % F);
    int b = (int)((idx / F) % B_);
    int t = (int)(idx / ((size_t)F * B_));
    int ol = (lengths[b] - 1) / 2 + 1;
    int ts = (t < ol) ? (ol - 1 - t) : (TC_ - 1 - (t - ol));
    out[idx] = in[((size_t)ts * B_ + b) * F + f];
}

// ---------------- block-tiled bf16 WMMA GEMM: 128x64 C-tile per block ----------------
// 8 waves = 8 M-tiles; B panel (64 cols x 32 k) staged in LDS with CDNA5 async
// global->LDS loads, double-buffered; A fragments register-direct per wave.
__global__ void ds2_gemm_wmma_kernel(const unsigned short* __restrict__ A,
                                     const unsigned short* __restrict__ Bt,
                                     float* __restrict__ C,
                                     int Mt, int Nt4, int Nt, int K,
                                     int lda, int ldb, int ldc) {
    __shared__ unsigned short sB[2][64 * 32];
    int tid = threadIdx.x;
    int wave = tid >> 5, lane = tid & 31;
    int mt8 = blockIdx.x / Nt4, nt4 = blockIdx.x % Nt4;
    int nt0 = nt4 * 4;
    int mt = mt8 * 8 + wave;
    bool mvalid = (mt < Mt);
    int mtc = mvalid ? mt : Mt - 1;
    const unsigned short* Ap = A + (size_t)mtc * 16 * lda;

    // staging assignment: thread -> (row, 16B segment) of the 64x32 B panel
    int srow = tid >> 2, sseg = tid & 3;
    int gn = nt0 * 16 + srow; if (gn > Nt * 16 - 1) gn = Nt * 16 - 1;
    const unsigned short* Bsrc = Bt + (size_t)gn * ldb + sseg * 8;
    unsigned lds0 = (unsigned)(size_t)(&sB[0][0]) + (unsigned)((srow * 32 + sseg * 8) * 2);
    const unsigned bufBytes = 64 * 32 * 2;

    // stage k=0 into buffer 0
    async_load_b128(lds0, (const void*)Bsrc);
    wait_async0();
    __syncthreads();

    v8f acc[4] = {{}, {}, {}, {}};
    int buf = 0;
    for (int k = 0; k < K; k += 32) {
        if (k + 32 < K)
            async_load_b128(lds0 + (unsigned)(buf ^ 1) * bufBytes, (const void*)(Bsrc + k + 32));
        __builtin_prefetch(Ap + k + 256, 0, 0);      // global_prefetch_b8 ahead on streamed A
        v16bf a = load_a_frag(Ap + k, lda, lane);
        const unsigned short* sBc = &sB[buf][0];
#pragma unroll
        for (int j = 0; j < 4; ++j) {
            v16bf b = load_b_frag(sBc + j * 16 * 32, 32, lane);
            acc[j] = __builtin_amdgcn_wmma_f32_16x16x32_bf16(false, a, false, b, (short)0, acc[j], false, false);
        }
        wait_async0();
        __syncthreads();
        buf ^= 1;
    }
    if (!mvalid) return;
    int n = lane & 15, half = lane >> 4;
#pragma unroll
    for (int j = 0; j < 4; ++j) {
        if (nt0 + j >= Nt) break;
        float* Cp = C + (size_t)mt * 16 * ldc + (size_t)(nt0 + j) * 16;
#pragma unroll
        for (int r = 0; r < 8; ++r) Cp[(size_t)(r + 8 * half) * ldc + n] = acc[j][r];
    }
}

// ---------------- persistent BiLSTM scan (both directions), LDS-resident Wh ----------------
__device__ __forceinline__ float lstm_elem(const float* sG, const float* xwt, const float* bias,
                                           int b, int u, int j0, float cprev,
                                           unsigned short* hcur, float* hseq, int t) {
    int j = j0 + u;
    float gi = sG[b * 128 + u]        + xwt[b * G4_ + j]            + bias[j];
    float gf = sG[b * 128 + 32 + u]   + xwt[b * G4_ + H_ + j]       + bias[H_ + j];
    float gg = sG[b * 128 + 64 + u]   + xwt[b * G4_ + 2 * H_ + j]   + bias[2 * H_ + j];
    float go = sG[b * 128 + 96 + u]   + xwt[b * G4_ + 3 * H_ + j]   + bias[3 * H_ + j];
    float c2 = sigmf(gf) * cprev + sigmf(gi) * tanhf(gg);
    float h2 = sigmf(go) * tanhf(c2);
    hcur[b * H_ + j] = f2bf(h2);
    hseq[((size_t)t * B_ + b) * H_ + j] = h2;
    return c2;
}

__global__ void ds2_lstm_scan_kernel(const float* __restrict__ xw_f, const float* __restrict__ xw_b,
                                     const float* __restrict__ bias_f, const float* __restrict__ bias_b,
                                     const unsigned short* __restrict__ WhT_f,
                                     const unsigned short* __restrict__ WhT_b,
                                     unsigned short* hcur_f, unsigned short* hcur_b,
                                     float* hseq_f, float* hseq_b,
                                     int* cntA, int* cntB, int T) {
    extern __shared__ char smem[];
    unsigned short* sWh = (unsigned short*)smem;    // 128 x 768 bf16  (196608 B)
    unsigned short* sH  = sWh + 128 * H_;           // 16 x 768 bf16   (24576 B)
    float* sG = (float*)(sH + B_ * H_);             // 16 x 128 f32    (8192 B)

    int wg = blockIdx.x;            // 0..23 : hidden slice
    int dir = blockIdx.y;           // 0 fwd / 1 bwd
    const float* xw   = dir ? xw_b  : xw_f;
    const float* bias = dir ? bias_b : bias_f;
    const unsigned short* WhT = dir ? WhT_b : WhT_f;
    unsigned short* hcur = dir ? hcur_b : hcur_f;
    float* hseq = dir ? hseq_b : hseq_f;
    volatile int* cA = (volatile int*)(cntA) + dir * T;
    volatile int* cB = (volatile int*)(cntB) + dir * T;

    int tid = threadIdx.x;
    int wave = tid >> 5, lane = tid & 31;
    int j0 = wg * 32;

    // Preload this WG's Wh slice: local row = g*32+u  <-  global col g*768 + j0 + u
    for (int row = 0; row < 128; ++row) {
        int g = row >> 5, u = row & 31;
        const unsigned int* s = (const unsigned int*)(WhT + ((size_t)(g * H_ + j0 + u)) * H_);
        unsigned int* d = (unsigned int*)(sWh + (size_t)row * H_);
        for (int i = tid; i < H_ / 2; i += blockDim.x) d[i] = s[i];
    }

    int b0 = tid & 15, u0 = tid >> 4;        // element 0: u in 0..15
    int b1 = b0,       u1 = u0 + 16;         // element 1: u in 16..31
    float c0 = 0.f, c1 = 0.f;
    __syncthreads();

    for (int t = 0; t < T; ++t) {
        // stage h(t-1) into LDS (zeros at t==0)
        {
            const unsigned int* s = (const unsigned int*)hcur;
            unsigned int* d = (unsigned int*)sH;
            if (t == 0) {
                for (int i = tid; i < B_ * H_ / 2; i += blockDim.x) d[i] = 0u;
            } else {
                for (int i = tid; i < B_ * H_ / 2; i += blockDim.x) d[i] = s[i];
            }
        }
        __syncthreads();
        // barrier A: all WGs of this direction finished staging h(t-1)
        if (tid == 0) {
            atomicAdd(&((int*)cA)[t], 1);
            while (cA[t] < NWG_) { __builtin_amdgcn_s_sleep(1); }
            __threadfence();
        }
        __syncthreads();

        // gates tile: wave computes 16 batch rows x 16 local gate cols from LDS
        v8f acc = {};
        const unsigned short* Bp = sWh + (size_t)wave * 16 * H_;
        for (int k = 0; k < H_; k += 32) {
            v16bf a = load_a_frag(sH + k, H_, lane);
            v16bf b = load_b_frag(Bp + k, H_, lane);
            acc = __builtin_amdgcn_wmma_f32_16x16x32_bf16(false, a, false, b, (short)0, acc, false, false);
        }
        {
            int n = lane & 15, half = lane >> 4;
#pragma unroll
            for (int r = 0; r < 8; ++r) sG[(r + 8 * half) * 128 + wave * 16 + n] = acc[r];
        }
        __syncthreads();

        // pointwise LSTM cell update (each thread owns 2 (b,u) elements; c stays in regs)
        const float* xwt = xw + (size_t)t * B_ * G4_;
        c0 = lstm_elem(sG, xwt, bias, b0, u0, j0, c0, hcur, hseq, t);
        c1 = lstm_elem(sG, xwt, bias, b1, u1, j0, c1, hcur, hseq, t);
        __syncthreads();
        // barrier B: all WGs wrote h(t)
        if (tid == 0) {
            __threadfence();
            atomicAdd(&((int*)cB)[t], 1);
            while (cB[t] < NWG_) { __builtin_amdgcn_s_sleep(1); }
            __threadfence();
        }
        __syncthreads();
    }
}

// ---------------- combine fwd + unflipped bwd ----------------
__global__ void ds2_combine_kernel(const float* __restrict__ hf, const float* __restrict__ hb,
                                   const int* __restrict__ lengths, float* __restrict__ out) {
    size_t idx = (size_t)blockIdx.x * blockDim.x + threadIdx.x;
    size_t total = (size_t)TC_ * B_ * H_;
    if (idx >= total) return;
    int h = (int)(idx % H_);
    int b = (int)((idx / H_) % B_);
    int t = (int)(idx / ((size_t)H_ * B_));
    int ol = (lengths[b] - 1) / 2 + 1;
    int ts = (t < ol) ? (ol - 1 - t) : (TC_ - 1 - (t - ol));
    out[idx] = hf[idx] + hb[((size_t)ts * B_ + b) * H_ + h];
}

// ---------------- eval-mode BN (+mask) -> bf16 (T,B,768) ----------------
__global__ void ds2_bn_eval_kernel(const float* __restrict__ x, const float* __restrict__ mean,
                                   const float* __restrict__ var, const float* __restrict__ scale,
                                   const float* __restrict__ bias, const int* __restrict__ lengths,
                                   unsigned short* __restrict__ out) {
    size_t idx = (size_t)blockIdx.x * blockDim.x + threadIdx.x;
    size_t total = (size_t)TC_ * B_ * H_;
    if (idx >= total) return;
    int h = (int)(idx % H_);
    int b = (int)((idx / H_) % B_);
    int t = (int)(idx / ((size_t)H_ * B_));
    int ol = (lengths[b] - 1) / 2 + 1;
    float v = 0.f;
    if (t < ol)
        v = (x[idx] - mean[h]) * rsqrtf(var[h] + 1e-5f) * scale[h] + bias[h];
    out[idx] = f2bf(v);
}

// ---------------- log_softmax over 29 classes + out_len tail ----------------
__global__ void ds2_logsoftmax_kernel(const float* __restrict__ logits, const int* __restrict__ lengths,
                                      float* __restrict__ out) {
    int row = blockIdx.x * blockDim.x + threadIdx.x;
    if (row < MROWS_) {
        const float* r = logits + (size_t)row * 32;
        float m = r[0];
        for (int k = 1; k < 29; ++k) m = fmaxf(m, r[k]);
        float s = 0.f;
        for (int k = 0; k < 29; ++k) s += expf(r[k] - m);
        float ls = m + logf(s);
        int b = row & 15, t = row >> 4;
        float* o = out + ((size_t)b * TC_ + t) * 29;
        for (int k = 0; k < 29; ++k) o[k] = r[k] - ls;
    }
    if (row < B_) out[(size_t)B_ * TC_ * 29 + row] = (float)((lengths[row] - 1) / 2 + 1);
}

// =====================================================================
extern "C" void kernel_launch(void* const* d_in, const int* in_sizes, int n_in,
                              void* d_out, int out_size, void* d_ws, size_t ws_size,
                              hipStream_t stream) {
    (void)in_sizes; (void)n_in; (void)out_size; (void)ws_size;
    const float* inputs   = (const float*)d_in[0];
    const int*   lengths  = (const int*)d_in[1];
    const float* conv1_W  = (const float*)d_in[2];
    const float* conv1_b  = (const float*)d_in[3];
    const float* conv2_W  = (const float*)d_in[4];
    const float* conv2_b  = (const float*)d_in[5];
    const float* cbn1_s   = (const float*)d_in[6];
    const float* cbn1_bi  = (const float*)d_in[7];
    const float* cbn2_s   = (const float*)d_in[8];
    const float* cbn2_bi  = (const float*)d_in[9];
    const float* l1f_Wi   = (const float*)d_in[10];
    const float* l1f_Wh   = (const float*)d_in[11];
    const float* l1f_b    = (const float*)d_in[12];
    const float* l1b_Wi   = (const float*)d_in[13];
    const float* l1b_Wh   = (const float*)d_in[14];
    const float* l1b_b    = (const float*)d_in[15];
    const float* lw_Wi_f  = (const float*)d_in[16];
    const float* lw_Wi_b  = (const float*)d_in[17];
    const float* lw_Wh_f  = (const float*)d_in[18];
    const float* lw_Wh_b  = (const float*)d_in[19];
    const float* lw_b_f   = (const float*)d_in[20];
    const float* lw_b_b   = (const float*)d_in[21];
    const float* rbn_s    = (const float*)d_in[22];
    const float* rbn_bi   = (const float*)d_in[23];
    const float* rbn_m    = (const float*)d_in[24];
    const float* rbn_v    = (const float*)d_in[25];
    const float* on_s     = (const float*)d_in[26];
    const float* on_bi    = (const float*)d_in[27];
    const float* on_m     = (const float*)d_in[28];
    const float* on_v     = (const float*)d_in[29];
    const float* fc_W     = (const float*)d_in[30];
    float* out = (float*)d_out;

    // ---- workspace carve-out ----
    char* p = (char*)d_ws;
    auto carve = [&](size_t bytes) -> void* {
        void* r = (void*)p;
        p += (bytes + 255) & ~(size_t)255;
        return r;
    };
    float*          conv1_out = (float*)carve((size_t)B_*D1_*TC_*C_*4);
    unsigned short* bn1_out   = (unsigned short*)carve((size_t)B_*D1_*TC_*C_*2);
    float*          conv2_out = (float*)carve((size_t)B_*D2_*TC_*C_*4);
    unsigned short* feats     = (unsigned short*)carve((size_t)MROWS_*F1_*2);
    unsigned short* feats_fl  = (unsigned short*)carve((size_t)MROWS_*F1_*2);
    float*          xw_f      = (float*)carve((size_t)MROWS_*G4_*4);
    float*          xw_b      = (float*)carve((size_t)MROWS_*G4_*4);
    float*          hseq_f    = (float*)carve((size_t)MROWS_*H_*4);
    float*          hseq_b    = (float*)carve((size_t)MROWS_*H_*4);
    float*          xcur      = (float*)carve((size_t)MROWS_*H_*4);
    unsigned short* hnorm     = (unsigned short*)carve((size_t)MROWS_*H_*2);
    unsigned short* hnorm_fl  = (unsigned short*)carve((size_t)MROWS_*H_*2);
    float*          logits    = (float*)carve((size_t)MROWS_*32*4);
    unsigned short* WiT1f     = (unsigned short*)carve((size_t)G4_*F1_*2);
    unsigned short* WiT1b     = (unsigned short*)carve((size_t)G4_*F1_*2);
    unsigned short* WiTf      = (unsigned short*)carve((size_t)4*G4_*H_*2);
    unsigned short* WiTb      = (unsigned short*)carve((size_t)4*G4_*H_*2);
    unsigned short* WhTf      = (unsigned short*)carve((size_t)5*G4_*H_*2);
    unsigned short* WhTb      = (unsigned short*)carve((size_t)5*G4_*H_*2);
    unsigned short* conv2Wt   = (unsigned short*)carve((size_t)32*K2_*2);
    unsigned short* fcT       = (unsigned short*)carve((size_t)32*H_*2);
    unsigned short* hcur_f    = (unsigned short*)carve((size_t)B_*H_*2);
    unsigned short* hcur_b    = (unsigned short*)carve((size_t)B_*H_*2);
    int*            cnt       = (int*)carve(6000 * 4);
    float*          stats     = (float*)carve(128 * 4);
    int*            sum_ol    = (int*)carve(4);

    float* stat1 = stats;
    float* stat2 = stats + 64;

    // ---- init (counters, stats, sum of output lengths) ----
    ds2_init_kernel<<<1, 256, 0, stream>>>(cnt, stats, sum_ol, lengths);

    // ---- pack weights to bf16, N-major ----
    auto packT = [&](const float* src, unsigned short* dst, int K, int N, int Npad) {
        size_t total = (size_t)Npad * K;
        ds2_packT_kernel<<<(unsigned)((total + 255) / 256), 256, 0, stream>>>(src, dst, K, N, Npad);
    };
    packT(l1f_Wi, WiT1f, F1_, G4_, G4_);
    packT(l1b_Wi, WiT1b, F1_, G4_, G4_);
    packT(l1f_Wh, WhTf + 0, H_, G4_, G4_);
    packT(l1b_Wh, WhTb + 0, H_, G4_, G4_);
    for (int i = 0; i < 4; ++i) {
        packT(lw_Wi_f + (size_t)i*H_*G4_, WiTf + (size_t)i*G4_*H_, H_, G4_, G4_);
        packT(lw_Wi_b + (size_t)i*H_*G4_, WiTb + (size_t)i*G4_*H_, H_, G4_, G4_);
        packT(lw_Wh_f + (size_t)i*H_*G4_, WhTf + (size_t)(i+1)*G4_*H_, H_, G4_, G4_);
        packT(lw_Wh_b + (size_t)i*H_*G4_, WhTb + (size_t)(i+1)*G4_*H_, H_, G4_, G4_);
    }
    packT(conv2_W, conv2Wt, K2_, 32, 32);
    packT(fc_W, fcT, H_, 29, 32);

    // ---- conv1 + masked BN1 ----
    {
        int total = B_*D1_*TC_*C_;
        ds2_conv1_kernel<<<(total + 255) / 256, 256, 0, stream>>>(inputs, conv1_W, conv1_b, conv1_out);
        ds2_bn_stats_kernel<<<2048, 256, 0, stream>>>(conv1_out, stat1, total);
        ds2_bn1_apply_kernel<<<(total + 255) / 256, 256, 0, stream>>>(
            conv1_out, stat1, cbn1_s, cbn1_bi, lengths, sum_ol, bn1_out);
    }
    // ---- conv2 (implicit GEMM WMMA, 16x32 per wave) + masked BN2 -> feats ----
    {
        int tiles = (B_*D2_*TC_) / 16;
        ds2_conv2_wmma_kernel<<<(tiles + 7) / 8, 256, 0, stream>>>(bn1_out, conv2Wt, conv2_b, conv2_out);
        int total = B_*D2_*TC_*C_;
        ds2_bn_stats_kernel<<<2048, 256, 0, stream>>>(conv2_out, stat2, total);
        ds2_bn2_apply_kernel<<<(total + 255) / 256, 256, 0, stream>>>(
            conv2_out, stat2, cbn2_s, cbn2_bi, lengths, sum_ol, feats);
    }

    const size_t lstmSmem = (size_t)(128*H_ + B_*H_) * 2 + (size_t)B_*128*4; // 229376 B
    auto gemm = [&](const unsigned short* A, const unsigned short* Bt, float* C,
                    int M, int N, int K) {
        int Mt = M / 16, Nt = N / 16, Nt4 = (Nt + 3) / 4;
        int mt8 = (Mt + 7) / 8;
        ds2_gemm_wmma_kernel<<<mt8 * Nt4, 256, 0, stream>>>(
            A, Bt, C, Mt, Nt4, Nt, K, K, K, N);
    };
    size_t totTBF = (size_t)TC_*B_*F1_;
    size_t totTBH = (size_t)TC_*B_*H_;

    // ---- layer 1 BiLSTM ----
    ds2_flip_kernel<<<(unsigned)((totTBF + 255)/256), 256, 0, stream>>>(feats, feats_fl, lengths, F1_);
    gemm(feats,    WiT1f, xw_f, MROWS_, G4_, F1_);
    gemm(feats_fl, WiT1b, xw_b, MROWS_, G4_, F1_);
    ds2_lstm_scan_kernel<<<dim3(NWG_, 2), 256, lstmSmem, stream>>>(
        xw_f, xw_b, l1f_b, l1b_b, WhTf, WhTb, hcur_f, hcur_b,
        hseq_f, hseq_b, cnt + 0*1200, cnt + 0*1200 + 600, TC_);
    ds2_combine_kernel<<<(unsigned)((totTBH + 255)/256), 256, 0, stream>>>(hseq_f, hseq_b, lengths, xcur);

    // ---- layers 2..5: eval BN + mask, BiLSTM ----
    for (int i = 0; i < 4; ++i) {
        ds2_bn_eval_kernel<<<(unsigned)((totTBH + 255)/256), 256, 0, stream>>>(
            xcur, rbn_m + i*H_, rbn_v + i*H_, rbn_s + i*H_, rbn_bi + i*H_, lengths, hnorm);
        ds2_flip_kernel<<<(unsigned)((totTBH + 255)/256), 256, 0, stream>>>(hnorm, hnorm_fl, lengths, H_);
        gemm(hnorm,    WiTf + (size_t)i*G4_*H_, xw_f, MROWS_, G4_, H_);
        gemm(hnorm_fl, WiTb + (size_t)i*G4_*H_, xw_b, MROWS_, G4_, H_);
        ds2_lstm_scan_kernel<<<dim3(NWG_, 2), 256, lstmSmem, stream>>>(
            xw_f, xw_b, lw_b_f + i*G4_, lw_b_b + i*G4_,
            WhTf + (size_t)(i+1)*G4_*H_, WhTb + (size_t)(i+1)*G4_*H_,
            hcur_f, hcur_b, hseq_f, hseq_b,
            cnt + (i+1)*1200, cnt + (i+1)*1200 + 600, TC_);
        ds2_combine_kernel<<<(unsigned)((totTBH + 255)/256), 256, 0, stream>>>(hseq_f, hseq_b, lengths, xcur);
    }

    // ---- output BN + FC + log_softmax ----
    ds2_bn_eval_kernel<<<(unsigned)((totTBH + 255)/256), 256, 0, stream>>>(
        xcur, on_m, on_v, on_s, on_bi, lengths, hnorm);
    gemm(hnorm, fcT, logits, MROWS_, 32, H_);
    ds2_logsoftmax_kernel<<<(MROWS_ + 255) / 256, 256, 0, stream>>>(logits, lengths, out);
}
